// QuantumAttention_21534966022685
// MI455X (gfx1250) — matrix operand order repaired and checked by reading, not compiled, not run on hardware
//
#include <hip/hip_runtime.h>
#include <hip/hip_bf16.h>
#include <math.h>

// Problem dims (fixed by reference)
#define BQ   4
#define SEQ  2048
#define DIM  1024

typedef __attribute__((ext_vector_type(16))) __bf16 v16bf;
typedef __attribute__((ext_vector_type(8)))  float  v8f;
typedef __attribute__((ext_vector_type(4)))  unsigned int v4u;
typedef __attribute__((ext_vector_type(8)))  int v8i;
typedef __attribute__((ext_vector_type(4)))  int v4i;

union Frag16 { unsigned int u[8]; v16bf v; };

__device__ __forceinline__ unsigned short f32_to_bf16(float f) {
  unsigned int u = __float_as_uint(f);
  unsigned int r = u + 0x7FFFu + ((u >> 16) & 1u);   // round-to-nearest-even
  return (unsigned short)(r >> 16);
}

// ---------------------------------------------------------------------
// CDNA5 async global->LDS copy (16B per lane), tracked by ASYNCcnt.
// Low 32 bits of a generic LDS pointer are the LDS byte offset.
// ---------------------------------------------------------------------
__device__ __forceinline__ void async_b128(unsigned lds_off, const void* gsrc) {
  asm volatile("global_load_async_to_lds_b128 %0, %1, off"
               :
               : "v"(lds_off), "v"(gsrc)
               : "memory");
}
__device__ __forceinline__ void wait_async0() {
  asm volatile("s_wait_asynccnt 0x0" ::: "memory");
}

// ---------------------------------------------------------------------
// Tensor Data Mover: one descriptor moves a whole 128x64-bf16 tile
// (128 rows x 128B, row stride K*2 bytes) global -> LDS. TENSORcnt.
// D# packing per CDNA5 ISA ch.8 (groups 0/1; 2D tile so groups 2/3 = 0).
// data_size = 3 (8-byte units): row = 16 units, stride = K/4 units.
// ---------------------------------------------------------------------
__device__ __forceinline__ void tdm_load_tile(const void* gsrc, unsigned lds_off,
                                              int K_elems) {
  const size_t ga = (size_t)gsrc;
  v4u g0;
  g0[0] = 1u;                                           // count=1, user desc
  g0[1] = lds_off;                                      // lds_addr (bytes)
  g0[2] = (unsigned)(ga & 0xFFFFFFFFu);                 // global_addr[31:0]
  g0[3] = (unsigned)((ga >> 32) & 0x01FFFFFFu)          // global_addr[56:32]
          | (2u << 30);                                 // type = 2 ("image")
  const unsigned data_size = 3u;                        // 8-byte units
  const unsigned td0 = 16u;                             // tensor_dim0 (units)
  const unsigned td1 = 128u;                            // tensor_dim1 (rows)
  const unsigned tile0 = 16u;                           // tile_dim0 (units)
  const unsigned tile1 = 128u;                          // tile_dim1 (rows)
  const unsigned stride0 = (unsigned)(K_elems >> 2);    // K*2B / 8B
  v8i g1;
  g1[0] = (int)(data_size << 16);                       // wg_mask=0, flags=0
  g1[1] = (int)((td0 & 0xFFFFu) << 16);                 // tensor_dim0[15:0]
  g1[2] = (int)((td0 >> 16) | ((td1 & 0xFFFFu) << 16)); // td0[31:16], td1[15:0]
  g1[3] = (int)((td1 >> 16) | (tile0 << 16));           // td1[31:16], tile_dim0
  g1[4] = (int)tile1;                                   // tile_dim1, tile_dim2=0
  g1[5] = (int)stride0;                                 // dim0_stride[31:0]
  g1[6] = 0;                                            // dim0_stride[47:32], dim1_stride[15:0]
  g1[7] = 0;                                            // dim1_stride[47:16]
  const v4i z4 = {0, 0, 0, 0};
#if defined(__clang_major__) && (__clang_major__ >= 23)
  const v8i z8 = {0, 0, 0, 0, 0, 0, 0, 0};
  __builtin_amdgcn_tensor_load_to_lds(g0, g1, z4, z4, z8, 0);
#else
  __builtin_amdgcn_tensor_load_to_lds(g0, g1, z4, z4, 0);
#endif
}

// ---- WMMA fragment loaders (k = 16*(lane>>4) + s for A and B, wave32) ----
// LDS tiles are [128 rows][64 bf16] (row stride 64).
__device__ __forceinline__ void load_a_frag(const unsigned short* lds, int m_off,
                                            int kk, int lane, unsigned int* au) {
  const unsigned short* p = lds + (m_off + (lane & 15)) * 64 + kk;
  const int g = (lane >> 4) << 3;
#pragma unroll
  for (int v = 0; v < 8; ++v) {
    const int kb = 2 * v + 2 * g;
    au[v] = *(const unsigned int*)(p + kb);
  }
}
__device__ __forceinline__ void load_b_frag(const unsigned short* lds, int n_off,
                                            int kk, int lane, unsigned int* bu) {
  const unsigned short* p =
      lds + (n_off + (lane & 15)) * 64 + kk + ((lane >> 4) << 4);
#pragma unroll
  for (int v = 0; v < 8; ++v) bu[v] = *(const unsigned int*)(p + 2 * v);
}

// Stage one 128x64 A tile (per-lane async b128, all 8 waves) and one
// 128x64 B tile (single TDM descriptor issued by wave 0).
__device__ __forceinline__ void stage_tiles(
    const unsigned short* __restrict__ A, const unsigned short* __restrict__ B,
    int K, int k0, int m0, int n0,
    unsigned short* ldsA, unsigned short* ldsB, int t) {
#pragma unroll
  for (int i = 0; i < 4; ++i) {
    const int linear = t + i * 256;          // 0..1023 chunks of 16B
    const int row = linear >> 3;             // 0..127
    const int c16 = (linear & 7) << 3;       // chunk offset in ushorts
    const unsigned short* g = A + (size_t)(m0 + row) * K + k0 + c16;
    const unsigned lo = (unsigned)(size_t)(ldsA + row * 64 + c16);
    async_b128(lo, g);
  }
  if ((t >> 5) == 0) {                       // wave 0 issues the TDM op
    const unsigned short* g = B + (size_t)n0 * K + k0;
    tdm_load_tile(g, (unsigned)(size_t)ldsB, K);
  }
}

// =====================================================================
// Unified double-buffered WMMA matmul.
//   C[M,N] = sum_seg A_seg[M,K] @ B_seg[N,K]^T   (both operands [row][k])
// Block tile 128x128, BK=64, 8 waves (4 M x 2 N), wave tile 32x64.
// EPI: 0 Q  : bf16 C0, +bias
//      1 K  : bf16 C0 +bias, and C1 = bf16(val/32)
//      2 V  : bf16 transposed per batch: C0[(b*DIM+col)*SEQ + s], +bias
//      3 RK : bf16 cos(pi*sigmoid(val+bias))
//      4 RQ : bf16 es*cos(pi*sigmoid(val+bias))
//      5 SC : f32 C0, no bias (scores)
//      6 PV : bf16 C0, no bias
//      7 OUT: f32 C0, +bias
// =====================================================================
template <int NSEG, int EPI>
__global__ __launch_bounds__(256) void mm_async_kernel(
    const unsigned short* __restrict__ A0,
    const unsigned short* __restrict__ B0,
    const unsigned short* __restrict__ A1,
    const unsigned short* __restrict__ B1,
    int K,
    const float* __restrict__ bias,
    const float* __restrict__ es_ptr,
    void* __restrict__ C0,
    void* __restrict__ C1,
    int ldc,
    size_t sAz, size_t sBz, size_t sCzBytes) {
  __shared__ __align__(16) unsigned short ldsA[2][128 * 64];
  __shared__ __align__(16) unsigned short ldsB[2][128 * 64];

  const int t    = threadIdx.x;
  const int lane = t & 31;
  const int wave = t >> 5;
  const int wm   = wave & 3;
  const int wn   = wave >> 2;
  const int m0   = blockIdx.y * 128;
  const int n0   = blockIdx.x * 128;
  const int z    = blockIdx.z;

  const unsigned short* As[2] = {A0 + (size_t)z * sAz,
                                 (NSEG > 1) ? (A1 + (size_t)z * sAz) : A0};
  const unsigned short* Bs[2] = {B0 + (size_t)z * sBz,
                                 (NSEG > 1) ? (B1 + (size_t)z * sBz) : B0};
  char* C0z = (char*)C0 + (size_t)z * sCzBytes;

  v8f acc[2][4];
#pragma unroll
  for (int i = 0; i < 2; ++i)
#pragma unroll
    for (int j = 0; j < 4; ++j)
#pragma unroll
      for (int v = 0; v < 8; ++v) acc[i][j][v] = 0.0f;

  const int steps_per_seg = K >> 6;
  const int total = NSEG * steps_per_seg;

  stage_tiles(As[0], Bs[0], K, 0, m0, n0, ldsA[0], ldsB[0], t);
  int buf = 0;

  for (int step = 0; step < total; ++step) {
    wait_async0();                                  // my A-tile copies landed
    if (wave == 0) __builtin_amdgcn_s_wait_tensorcnt(0);  // B-tile TDM landed
    __syncthreads();                                // everyone's did
    if (step + 1 < total) {
      const int ns = (step + 1) / steps_per_seg;
      const int nk = ((step + 1) % steps_per_seg) << 6;
      stage_tiles(As[ns], Bs[ns], K, nk, m0, n0,
                  ldsA[buf ^ 1], ldsB[buf ^ 1], t); // overlaps wmma below
    }
    const unsigned short* tA = ldsA[buf];
    const unsigned short* tB = ldsB[buf];
#pragma unroll
    for (int kk = 0; kk < 64; kk += 32) {
      Frag16 a[2], b[4];
#pragma unroll
      for (int i = 0; i < 2; ++i) load_a_frag(tA, wm * 32 + i * 16, kk, lane, a[i].u);
#pragma unroll
      for (int j = 0; j < 4; ++j) load_b_frag(tB, wn * 64 + j * 16, kk, lane, b[j].u);
#pragma unroll
      for (int i = 0; i < 2; ++i)
#pragma unroll
        for (int j = 0; j < 4; ++j)
          acc[i][j] = __builtin_amdgcn_wmma_f32_16x16x32_bf16(
              false, a[i].v, false, b[j].v, (short)0, acc[i][j], false, false);
    }
    buf ^= 1;
  }

  // Epilogue. C/D layout: lane&15 = N, VGPR v -> M = v + 8*(lane>>4).
  float esv = 1.0f;
  if constexpr (EPI == 4) esv = es_ptr[0];
  const int g  = lane >> 4;
  const int ln = lane & 15;
#pragma unroll
  for (int i = 0; i < 2; ++i) {
#pragma unroll
    for (int j = 0; j < 4; ++j) {
      const int col = n0 + wn * 64 + j * 16 + ln;
      float bv = 0.0f;
      if constexpr (EPI != 5 && EPI != 6) bv = bias[col];
#pragma unroll
      for (int v = 0; v < 8; ++v) {
        const int row = m0 + wm * 32 + i * 16 + v + 8 * g;
        float val = acc[i][j][v] + bv;
        if constexpr (EPI == 3 || EPI == 4) {
          const float s = 1.0f / (1.0f + __expf(-val));
          val = __cosf(3.14159265358979f * s);     // cos(2*theta)
          if constexpr (EPI == 4) val *= esv;
        }
        if constexpr (EPI == 5 || EPI == 7) {
          ((float*)C0z)[(size_t)row * ldc + col] = val;
        } else if constexpr (EPI == 2) {
          const int b = row >> 11;                 // /SEQ
          const int s = row & (SEQ - 1);
          ((unsigned short*)C0z)[((size_t)b * DIM + col) * SEQ + s] =
              f32_to_bf16(val);
        } else {
          ((unsigned short*)C0z)[(size_t)row * ldc + col] = f32_to_bf16(val);
        }
        if constexpr (EPI == 1) {
          ((unsigned short*)C1)[(size_t)row * ldc + col] =
              f32_to_bf16(val * 0.03125f);         // K/sqrt(D)
        }
      }
    }
  }
}

// =====================================================================
// Row softmax over SEQ=2048, in place; also writes bf16 copy for P@V.
// =====================================================================
__global__ __launch_bounds__(256) void softmax_kernel(
    float* __restrict__ attn, unsigned short* __restrict__ Pb) {
  const size_t rb = (size_t)blockIdx.x * SEQ;
  float* p = attn + rb;
  unsigned short* pb = Pb + rb;
  __shared__ float red[256];
  const int t = threadIdx.x;

  float vals[8];
  float mx = -3.0e38f;
#pragma unroll
  for (int i = 0; i < 8; ++i) {
    vals[i] = p[t + i * 256];
    mx = fmaxf(mx, vals[i]);
  }
  red[t] = mx;
  __syncthreads();
  for (int s = 128; s > 0; s >>= 1) {
    if (t < s) red[t] = fmaxf(red[t], red[t + s]);
    __syncthreads();
  }
  mx = red[0];
  __syncthreads();

  float sum = 0.0f;
#pragma unroll
  for (int i = 0; i < 8; ++i) {
    vals[i] = __expf(vals[i] - mx);
    sum += vals[i];
  }
  red[t] = sum;
  __syncthreads();
  for (int s = 128; s > 0; s >>= 1) {
    if (t < s) red[t] += red[t + s];
    __syncthreads();
  }
  const float inv = 1.0f / red[0];
#pragma unroll
  for (int i = 0; i < 8; ++i) {
    const float o = vals[i] * inv;
    p[t + i * 256] = o;
    pb[t + i * 256] = f32_to_bf16(o);
  }
}

// ---------------------------------------------------------------------
__global__ __launch_bounds__(256) void cvt_f32_bf16_kernel(
    const float* __restrict__ in, unsigned short* __restrict__ out, int n) {
  const int i = blockIdx.x * 256 + threadIdx.x;
  if (i < n) out[i] = f32_to_bf16(in[i]);
}

// f32 [K,N] -> bf16 [N,K] (weights pre-transposed so GEMM B-staging is a
// contiguous copy). 32x32 LDS tile transpose, 256 threads.
__global__ __launch_bounds__(256) void cvt_transpose_kernel(
    const float* __restrict__ in, unsigned short* __restrict__ out) {
  __shared__ float tile[32][33];
  const int bx = blockIdx.x * 32;   // N tile
  const int by = blockIdx.y * 32;   // K tile
  const int tx = threadIdx.x & 31;
  const int ty = threadIdx.x >> 5;  // 0..7
#pragma unroll
  for (int i = 0; i < 32; i += 8)
    tile[ty + i][tx] = in[(size_t)(by + ty + i) * DIM + bx + tx];
  __syncthreads();
#pragma unroll
  for (int i = 0; i < 32; i += 8)
    out[(size_t)(bx + ty + i) * DIM + by + tx] = f32_to_bf16(tile[tx][ty + i]);
}

// =====================================================================
extern "C" void kernel_launch(void* const* d_in, const int* in_sizes, int n_in,
                              void* d_out, int out_size, void* d_ws, size_t ws_size,
                              hipStream_t stream) {
  const float* x  = (const float*)d_in[0];
  const float* Wq = (const float*)d_in[1];
  const float* bq = (const float*)d_in[2];
  const float* Wk = (const float*)d_in[3];
  const float* bk = (const float*)d_in[4];
  const float* Wv = (const float*)d_in[5];
  const float* bv = (const float*)d_in[6];
  const float* Wt = (const float*)d_in[7];
  const float* bt = (const float*)d_in[8];
  const float* Wo = (const float*)d_in[9];
  const float* bo = (const float*)d_in[10];
  const float* es = (const float*)d_in[11];

  float* out  = (float*)d_out;                       // [B,S,D]
  float* attn = out + (size_t)BQ * SEQ * DIM;        // [B,S,S]

  const size_t ND = (size_t)BQ * SEQ * DIM;          // 8,388,608
  const size_t NW = (size_t)DIM * DIM;               // 1,048,576
  const size_t NP = (size_t)BQ * SEQ * SEQ;          // 16,777,216

  // Workspace carve-up (bf16 buffers), 256B aligned.
  char* ws = (char*)d_ws;
  size_t off = 0;
  auto carve = [&](size_t elems) {
    void* p = ws + off;
    off = (off + elems * sizeof(unsigned short) + 255) & ~(size_t)255;
    return (unsigned short*)p;
  };
  unsigned short* xb  = carve(ND);   // x, bf16
  unsigned short* WqT = carve(NW);   // weights, bf16, pre-transposed [N][K]
  unsigned short* WkT = carve(NW);
  unsigned short* WvT = carve(NW);
  unsigned short* WtT = carve(NW);
  unsigned short* WoT = carve(NW);
  unsigned short* Qb  = carve(ND);   // Q (unscaled, feeds theta GEMM)
  unsigned short* Kb  = carve(ND);   // K (unscaled, feeds theta GEMM)
  unsigned short* Ks  = carve(ND);   // K / sqrt(D)
  unsigned short* Vt  = carve(ND);   // V transposed per batch [b][d][s]
  unsigned short* RQ  = carve(ND);   // es * cos(pi*sigmoid(Q@Wt+bt))
  unsigned short* RK  = carve(ND);   // cos(pi*sigmoid(K@Wt+bt))
  unsigned short* Pb  = carve(NP);   // softmax probs, bf16
  unsigned short* Ob  = carve(ND);   // (P@V), bf16
  (void)ws_size; (void)in_sizes; (void)n_in; (void)out_size;

  // 1) conversions (x flat; weights transposed)
  cvt_f32_bf16_kernel<<<(unsigned)((ND + 255) / 256), 256, 0, stream>>>(x, xb, (int)ND);
  {
    dim3 gT(DIM / 32, DIM / 32);
    cvt_transpose_kernel<<<gT, 256, 0, stream>>>(Wq, WqT);
    cvt_transpose_kernel<<<gT, 256, 0, stream>>>(Wk, WkT);
    cvt_transpose_kernel<<<gT, 256, 0, stream>>>(Wv, WvT);
    cvt_transpose_kernel<<<gT, 256, 0, stream>>>(Wt, WtT);
    cvt_transpose_kernel<<<gT, 256, 0, stream>>>(Wo, WoT);
  }

  const int M = BQ * SEQ;                            // 8192
  dim3 gProj(DIM / 128, M / 128, 1);                 // 8 x 64
  // 2) projections
  mm_async_kernel<1, 0><<<gProj, 256, 0, stream>>>(
      xb, WqT, nullptr, nullptr, DIM, bq, nullptr, Qb, nullptr, DIM, 0, 0, 0);
  mm_async_kernel<1, 1><<<gProj, 256, 0, stream>>>(
      xb, WkT, nullptr, nullptr, DIM, bk, nullptr, Kb, Ks, DIM, 0, 0, 0);
  mm_async_kernel<1, 2><<<gProj, 256, 0, stream>>>(
      xb, WvT, nullptr, nullptr, DIM, bv, nullptr, Vt, nullptr, DIM, 0, 0, 0);
  // 3) theta GEMMs
  mm_async_kernel<1, 3><<<gProj, 256, 0, stream>>>(
      Kb, WtT, nullptr, nullptr, DIM, bt, nullptr, RK, nullptr, DIM, 0, 0, 0);
  mm_async_kernel<1, 4><<<gProj, 256, 0, stream>>>(
      Qb, WtT, nullptr, nullptr, DIM, bt, es, RQ, nullptr, DIM, 0, 0, 0);
  // 4) scores: Q·(K/32) + (es·r_q)·r_k, f32 into attn region
  dim3 gSc(SEQ / 128, SEQ / 128, BQ);                // 16 x 16 x 4
  mm_async_kernel<2, 5><<<gSc, 256, 0, stream>>>(
      Qb, Ks, RQ, RK, DIM, nullptr, nullptr, attn, nullptr, SEQ,
      (size_t)SEQ * DIM, (size_t)SEQ * DIM, (size_t)SEQ * SEQ * sizeof(float));
  // 5) softmax rows (f32 in place + bf16 copy)
  softmax_kernel<<<BQ * SEQ, 256, 0, stream>>>(attn, Pb);
  // 6) O = P @ V via Vt
  dim3 gPV(DIM / 128, SEQ / 128, BQ);                // 8 x 16 x 4
  mm_async_kernel<1, 6><<<gPV, 256, 0, stream>>>(
      Pb, Vt, nullptr, nullptr, SEQ, nullptr, nullptr, Ob, nullptr, DIM,
      (size_t)SEQ * SEQ, (size_t)DIM * SEQ, (size_t)SEQ * DIM * sizeof(unsigned short));
  // 7) out = O @ Wo + bo (f32)
  mm_async_kernel<1, 7><<<gProj, 256, 0, stream>>>(
      Ob, WoT, nullptr, nullptr, DIM, bo, nullptr, out, nullptr, DIM, 0, 0, 0);
}
